// AdditiveAttention_33569464385999
// MI455X (gfx1250) — matrix operand order repaired and checked by reading, not compiled, not run on hardware
//
#include <hip/hip_runtime.h>
#include <math.h>

typedef __attribute__((ext_vector_type(16))) __bf16 v16bf;
typedef __attribute__((ext_vector_type(8)))  __bf16 v8bf;
typedef __attribute__((ext_vector_type(8)))  float  v8f;

namespace {
constexpr int kB = 4096;
constexpr int kS = 200;   // candidates
constexpr int kD = 256;   // candidate dim (K of GEMM)
constexpr int kQ = 200;   // query dim (N of GEMM)

constexpr int kMT   = 13;   // ceil(200/16) M tiles
constexpr int kNT   = 13;   // ceil(200/16) N tiles
constexpr int kKS   = 8;    // 256 / 32 K-steps per tile
constexpr int kWROW = 264;  // padded LDS row (bf16): 256+8 -> 132 dwords, 4-bank skew
constexpr int kNPAD = 208;  // padded M/N extent
}

// Branch-free tanh: one v_exp_f32 (TRANS pipe, co-executes with XDL WMMA),
// one v_rcp_f32, no exec-mask slow paths. Clamp keeps e finite (no inf/inf).
__device__ __forceinline__ float fast_tanh(float x) {
  const float xc = fminf(fmaxf(x, -15.f), 15.f);
  const float e  = __expf(2.f * xc);
  return (e - 1.f) * __builtin_amdgcn_rcpf(e + 1.f);
}

__global__ __launch_bounds__(256, 1)
void addattn_fused_kernel(const float* __restrict__ cand,
                          const float* __restrict__ W,
                          const float* __restrict__ bias,
                          const float* __restrict__ q,
                          float* __restrict__ out)
{
  __shared__ __bf16 sWt[kNPAD * kWROW];   // W^T in bf16 (~109.8 KB)
  __shared__ float  sBias[kNPAD];
  __shared__ float  sQ[kNPAD];
  __shared__ float  sScore[kNPAD];
  __shared__ float  sWeight[kNPAD];

  const int tid  = threadIdx.x;
  const int lane = tid & 31;
  const int wave = tid >> 5;
  const int half = lane >> 4;   // which 16-lane half of the wave
  const int l15  = lane & 15;
  const int bidx = blockIdx.x;
  const float* candB = cand + (size_t)bidx * kS * kD;

  // ---- Phase 0: stage W^T (bf16), bias, q into LDS ----
  // W is [D=256][Q=200] row-major; store transposed: sWt[n][k], K contiguous.
  for (int idx = tid; idx < kNPAD * kD; idx += 256) {
    const int n = idx % kNPAD;          // consecutive tids -> consecutive n (coalesced)
    const int k = idx / kNPAD;
    const float v = (n < kQ) ? W[(size_t)k * kQ + n] : 0.0f;
    sWt[n * kWROW + k] = (__bf16)v;
  }
  if (tid < kNPAD) {
    sBias[tid] = (tid < kQ) ? bias[tid] : 0.0f;
    sQ[tid]    = (tid < kQ) ? q[tid]    : 0.0f;
  }
  __syncthreads();

  // ---- Phase 1: scores[m] = sum_n tanh((cand@W)[m][n] + bias[n]) * q[n] ----
  for (int mt = wave; mt < kMT; mt += 8) {
    const int mbase = mt * 16;
    const int arow  = mbase + l15;
    const bool rowOK = (arow < kS);
    const float* aptr = candB + (size_t)arow * kD;

    // A fragments for the whole K extent (16-bit A 16x32 layout:
    // lane half 0: K = kb..kb+7 and kb+16..kb+23 with kb = 32*kk;
    // lane half 1: same with kb = 32*kk + 8).
    v16bf afrag[kKS];
#pragma unroll
    for (int kk = 0; kk < kKS; ++kk) {
      const int kb = kk * 32 + half * 8;
      float4 f0, f1, f2, f3;
      if (rowOK) {
        f0 = *(const float4*)(aptr + kb);
        f1 = *(const float4*)(aptr + kb + 4);
        f2 = *(const float4*)(aptr + kb + 16);
        f3 = *(const float4*)(aptr + kb + 20);
      } else {
        f0 = f1 = f2 = f3 = make_float4(0.f, 0.f, 0.f, 0.f);
      }
      v16bf a;
      a[0]=(__bf16)f0.x;  a[1]=(__bf16)f0.y;  a[2]=(__bf16)f0.z;  a[3]=(__bf16)f0.w;
      a[4]=(__bf16)f1.x;  a[5]=(__bf16)f1.y;  a[6]=(__bf16)f1.z;  a[7]=(__bf16)f1.w;
      a[8]=(__bf16)f2.x;  a[9]=(__bf16)f2.y;  a[10]=(__bf16)f2.z; a[11]=(__bf16)f2.w;
      a[12]=(__bf16)f3.x; a[13]=(__bf16)f3.y; a[14]=(__bf16)f3.z; a[15]=(__bf16)f3.w;
      afrag[kk] = a;
    }

    float srow[8] = {0.f,0.f,0.f,0.f,0.f,0.f,0.f,0.f};

    for (int nt = 0; nt < kNT; ++nt) {
      const int nbase = nt * 16;
      v8f acc = {};
      // B fragment: column n = nbase + l15; lane half h takes K = 8h..8h+7
      // and 16+8h..16+8h+7 (same K order as A).
      const __bf16* bbase = &sWt[(nbase + l15) * kWROW + half * 8];
#pragma unroll
      for (int kk = 0; kk < kKS; ++kk) {
        const __bf16* bp = bbase + kk * 32;
        v8bf lo = *(const v8bf*)(bp);
        v8bf hi = *(const v8bf*)(bp + 16);
        v16bf bfrag = __builtin_shufflevector(lo, hi,
            0,1,2,3,4,5,6,7,8,9,10,11,12,13,14,15);
        acc = __builtin_amdgcn_wmma_f32_16x16x32_bf16(
            false, afrag[kk], false, bfrag, (short)0, acc, false, false);
      }
      // Fused epilogue: C/D layout -> lane holds col n = nbase+l15,
      // rows m = mbase + r + 8*half for r = 0..7.
      const int ncol = nbase + l15;
      const float bb = sBias[ncol];
      const float qq = sQ[ncol];
#pragma unroll
      for (int r = 0; r < 8; ++r) {
        srow[r] += fast_tanh(acc[r] + bb) * qq;
      }
    }

    // Reduce srow[r] across the 16 lanes of this half-wave (xor masks < 16
    // keep the reduction inside each 16-lane group of the wave32).
#pragma unroll
    for (int r = 0; r < 8; ++r) {
      float v = srow[r];
      v += __shfl_xor(v, 1, 32);
      v += __shfl_xor(v, 2, 32);
      v += __shfl_xor(v, 4, 32);
      v += __shfl_xor(v, 8, 32);
      if (l15 == 0) {
        const int m = mbase + half * 8 + r;
        if (m < kS) sScore[m] = v;
      }
    }
  }
  __syncthreads();

  // ---- Phase 2: softmax over S (LDS broadcast reads, done redundantly) ----
  float mx = -3.4e38f;
  for (int s = 0; s < kS; ++s) mx = fmaxf(mx, sScore[s]);
  if (tid < kS) sWeight[tid] = __expf(sScore[tid] - mx);
  __syncthreads();
  float denom = 0.f;
  for (int s = 0; s < kS; ++s) denom += sWeight[s];
  const float inv = __builtin_amdgcn_rcpf(denom);

  // ---- Phase 3: out[b][d] = (1/denom) * sum_s w[s] * cand[b][s][d] ----
  // 256 threads == D; loads are fully coalesced and L2-hot from phase 1.
  {
    const int d = tid;
    float accv = 0.f;
    for (int s = 0; s < kS; ++s) {
      accv += sWeight[s] * candB[(size_t)s * kD + d];
    }
    out[(size_t)bidx * kD + d] = accv * inv;
  }
}

extern "C" void kernel_launch(void* const* d_in, const int* in_sizes, int n_in,
                              void* d_out, int out_size, void* d_ws, size_t ws_size,
                              hipStream_t stream) {
  const float* cand = (const float*)d_in[0];   // [4096,200,256] f32
  const float* W    = (const float*)d_in[1];   // [256,200] f32
  const float* bias = (const float*)d_in[2];   // [200] f32
  const float* q    = (const float*)d_in[3];   // [200,1] f32
  float* out = (float*)d_out;                  // [4096,256] f32
  (void)in_sizes; (void)n_in; (void)out_size; (void)d_ws; (void)ws_size;

  addattn_fused_kernel<<<dim3(kB), dim3(256), 0, stream>>>(cand, W, bias, q, out);
}
